// CausalSelfAttention_10728828306076
// MI455X (gfx1250) — compile-verified, matrix operand order checked by reading, not run on hardware
//
#include <hip/hip_runtime.h>

typedef __bf16 bf16;
typedef __attribute__((ext_vector_type(16))) __bf16 v16bf;
typedef __attribute__((ext_vector_type(8)))  __bf16 v8bf;
typedef __attribute__((ext_vector_type(4)))  __bf16 v4bf;
typedef __attribute__((ext_vector_type(8)))  float  v8f;
typedef __attribute__((ext_vector_type(4)))  unsigned uint32x4;
typedef __attribute__((ext_vector_type(8)))  int      int32x8;

union F16frag { v16bf v; v8bf h[2]; };
union Pack2   { bf16 h[2]; unsigned u; };

#define Bsz 4
#define Tlen 2048
#define Cdim 768
#define Hn 12
#define Dh 64

// ---------------------------------------------------------------------------
// CDNA5 async global->LDS staging (ASYNCcnt path).
// LDS operand: low 32 bits of the generic pointer are the wave-relative LDS
// byte address (flat LDS aperture truncates high bits per ISA 10.2).
// ---------------------------------------------------------------------------
__device__ __forceinline__ void async_b128(const void* lds_ptr, const void* gptr) {
    asm volatile("global_load_async_to_lds_b128 %0, %1, off"
                 :: "v"((unsigned)(size_t)lds_ptr),
                    "v"((unsigned long long)(size_t)gptr)
                 : "memory");
}
#if __has_builtin(__builtin_amdgcn_s_wait_asynccnt)
#define WAIT_ASYNC(n) __builtin_amdgcn_s_wait_asynccnt(n)
#else
#define WAIT_ASYNC(n) asm volatile("s_wait_asynccnt %0" :: "i"(n) : "memory")
#endif
#if __has_builtin(__builtin_amdgcn_s_wait_tensorcnt)
#define WAIT_TENSOR(n) __builtin_amdgcn_s_wait_tensorcnt(n)
#else
#define WAIT_TENSOR(n) asm volatile("s_wait_tensorcnt %0" :: "i"(n) : "memory")
#endif

// ---------------------------------------------------------------------------
// CDNA5 Tensor Data Mover: one 2D tile global->LDS per instruction (TENSORcnt).
// D# built per ISA 8.3/8.4: group0 = {count/type, lds_addr, global_addr},
// group1 = {data_size/pad, tensor dims, tile dims, dim0 stride}.
// 2-group form (VADDR2/3 = NULL) covers 2D tiles. Issue from ONE wave only:
// TDM ignores EXEC, so the caller must gate with a *scalar* branch.
// pad_interval/pad_amount encode LDS row padding (payload 2^(pi+1) DWORDs,
// then pa+1 DWORDs skipped) to preserve bank-conflict-free row strides.
// ---------------------------------------------------------------------------
__device__ __forceinline__ void tdm_load_2d(const void* lds_ptr, const void* gptr,
                                            unsigned tile_x, unsigned tile_y,
                                            unsigned stride_elems,
                                            unsigned pad_interval, unsigned pad_amount) {
    const unsigned long long ga = (unsigned long long)(size_t)gptr;
    uint32x4 g0;
    g0[0] = 1u;                                            // count=1 (valid user D#)
    g0[1] = (unsigned)(size_t)lds_ptr;                     // lds_addr (bytes)
    g0[2] = (unsigned)ga;                                  // global_addr[31:0]
    g0[3] = ((unsigned)(ga >> 32) & 0x01FFFFFFu) | (2u << 30);  // addr[56:32] | type=2
    int32x8 g1;
    unsigned d0 = (1u << 16)                               // data_size = 2 bytes
                | (1u << 20)                               // pad_enable
                | (pad_interval << 22) | (pad_amount << 25);
    g1[0] = (int)d0;
    g1[1] = (int)((tile_x & 0xFFFFu) << 16);               // tensor_dim0[15:0]
    g1[2] = (int)(((tile_x >> 16) & 0xFFFFu) | ((tile_y & 0xFFFFu) << 16)); // td0 hi | td1 lo
    g1[3] = (int)(((tile_y >> 16) & 0xFFFFu) | ((tile_x & 0xFFFFu) << 16)); // td1 hi | tile_dim0
    g1[4] = (int)(tile_y & 0xFFFFu);                       // tile_dim1 | tile_dim2=0
    g1[5] = (int)stride_elems;                             // tensor_dim0_stride lo
    g1[6] = 0;                                             // stride hi | dim1_stride lo (2D)
    g1[7] = 0;
    asm volatile("tensor_load_to_lds %0, %1" :: "s"(g0), "s"(g1) : "memory");
}

// ---------------------------------------------------------------------------
// f32 -> bf16 conversion, 4 elems/thread
// ---------------------------------------------------------------------------
__global__ __launch_bounds__(256) void cvt_bf16_kernel(const float* __restrict__ in,
                                                       bf16* __restrict__ out, int n4) {
    int i = (blockIdx.x * 256 + threadIdx.x);
    if (i >= n4) return;
    const float4 f = ((const float4*)in)[i];
    bf16 o[4];
    o[0] = (bf16)f.x; o[1] = (bf16)f.y; o[2] = (bf16)f.z; o[3] = (bf16)f.w;
    *(uint2*)(out + (size_t)i * 4) = *(const uint2*)o;
}

// ---------------------------------------------------------------------------
// GEMM: C[M,N] = A[M,K](bf16) * B[K,N](bf16) + bias[N](f32)
// Block tile 128x128, BK=32. 8 waves in 4(M)x2(N) grid; wave tile 32x64.
// A tile: double-buffered LDS, staged by the Tensor Data Mover (one
// tensor_load_to_lds per tile, issued by wave 0, pipelined one tile ahead,
// LDS rows padded 64B+16B by the TDM pad feature).
// B tile: regs -> transposed packed-b32 stores.
// ---------------------------------------------------------------------------
template <bool OUT_BF16>
__global__ __launch_bounds__(256)
void gemm_bf16_kernel(const bf16* __restrict__ A, const bf16* __restrict__ Bm,
                      const float* __restrict__ bias, void* __restrict__ Cout,
                      int M, int N, int K) {
    __shared__ __align__(16) bf16 As[2][128][40];   // [buf][m][k] padded (TDM pad)
    __shared__ __align__(16) bf16 Bs[128][40];      // [n][k] (transposed) padded

    const int tid  = threadIdx.x;
    const int lane = tid & 31;
    const int wave = tid >> 5;     // 0..7
    const int wm   = wave >> 1;    // 0..3
    const int wn   = wave & 1;     // 0..1
    const int hi   = lane >> 4;    // 0/1
    const int l15  = lane & 15;

    const int nTilesN = N >> 7;
    const int bm = (blockIdx.x / nTilesN) << 7;
    const int bn = (blockIdx.x % nTilesN) << 7;

    // B staging map: row-pairs so transposed stores pack into b32
    const int brp = tid >> 4;            // 0..15 -> k rows 2*brp, 2*brp+1
    const int bc8 = (tid & 15) << 3;     // 0..120

    v8f acc[2][4];
    #pragma unroll
    for (int i = 0; i < 2; ++i)
        #pragma unroll
        for (int j = 0; j < 4; ++j) acc[i][j] = {};

    // scalar (SGPR) branch: only wave 0 issues TDM ops (TDM ignores EXEC)
    const bool issuer = (__builtin_amdgcn_readfirstlane(wave) == 0);
    const bf16* aTile = A + (size_t)bm * K;   // tile row base, col advances by k0

    // prologue: TDM-stage k-tile 0 into buffer 0
    if (issuer)
        tdm_load_2d(&As[0][0][0], aTile, /*x*/32, /*y*/128, /*stride*/K,
                    /*pad: 16 dwords payload*/3, /*4 dwords pad*/3);

    const int NK = K >> 5;
    for (int it = 0; it < NK; ++it) {
        const int k0  = it << 5;
        const int buf = it & 1;

        // prefetch B tile into regs (overlaps previous tile's compute)
        const bf16* bg0 = Bm + (size_t)(k0 + 2 * brp) * N + bn + bc8;
        v8bf b0 = *(const v8bf*)bg0;
        v8bf b1 = *(const v8bf*)(bg0 + N);

        __syncthreads();   // previous tile's readers done: Bs and As[buf^1] free

        if (issuer && it + 1 < NK)  // pipeline: TDM-stage next A tile
            tdm_load_2d(&As[buf ^ 1][0][0], aTile + (it + 1) * 32, 32, 128, K, 3, 3);

        // transposed B stores, packed pairs (k, k+1) -> one b32
        #pragma unroll
        for (int j = 0; j < 8; ++j) {
            Pack2 p; p.h[0] = b0[j]; p.h[1] = b1[j];
            *(unsigned*)&Bs[bc8 + j][2 * brp] = p.u;
        }

        if (issuer) {                      // TENSORcnt in-order: <=1 retires older
            if (it + 1 < NK) WAIT_TENSOR(1);
            else             WAIT_TENSOR(0);
        }
        __syncthreads();

        F16frag afr[2], bfr[4];
        #pragma unroll
        for (int i = 0; i < 2; ++i) {
            const bf16* p = &As[buf][wm * 32 + i * 16 + l15][0];
            afr[i].h[0] = *(const v8bf*)(p + hi * 8);
            afr[i].h[1] = *(const v8bf*)(p + 16 + hi * 8);
        }
        #pragma unroll
        for (int j = 0; j < 4; ++j) {
            const bf16* p = &Bs[wn * 64 + j * 16 + l15][0];
            bfr[j].h[0] = *(const v8bf*)(p + hi * 8);
            bfr[j].h[1] = *(const v8bf*)(p + 16 + hi * 8);
        }
        #pragma unroll
        for (int i = 0; i < 2; ++i)
            #pragma unroll
            for (int j = 0; j < 4; ++j)
                acc[i][j] = __builtin_amdgcn_wmma_f32_16x16x32_bf16(
                    false, afr[i].v, false, bfr[j].v, (short)0, acc[i][j], false, false);
    }

    // epilogue: C-layout row = v + hi*8 (per tile), col = l15
    #pragma unroll
    for (int i = 0; i < 2; ++i) {
        const int mbase = bm + wm * 32 + i * 16 + hi * 8;
        #pragma unroll
        for (int j = 0; j < 4; ++j) {
            const int n = bn + wn * 64 + j * 16 + l15;
            const float bv = bias[n];
            #pragma unroll
            for (int v = 0; v < 8; ++v) {
                const float val = acc[i][j][v] + bv;
                const size_t off = (size_t)(mbase + v) * N + n;
                if (OUT_BF16) ((bf16*)Cout)[off] = (bf16)val;
                else          ((float*)Cout)[off] = val;
            }
        }
    }
}

// ---------------------------------------------------------------------------
// Flash attention: one block per (b, h, 128-query tile); wave owns 16 q rows.
// K tile: double-buffered async-to-LDS (ASYNCcnt), pipelined one tile ahead.
// V tile: regs -> transposed [d][kt] with packed-b32 stores.
// ---------------------------------------------------------------------------
__global__ __launch_bounds__(256)
void attn_kernel(const bf16* __restrict__ qkv, bf16* __restrict__ yout) {
    __shared__ __align__(16) bf16 Klds[2][32][72];    // [buf][kt][d] padded
    __shared__ __align__(16) bf16 Vlds[64][40];       // [d][kt] padded (transposed)
    __shared__ __align__(16) bf16 Plds[8][16][32];    // per-wave P tile

    const int tid  = threadIdx.x;
    const int lane = tid & 31;
    const int wave = tid >> 5;
    const int hi   = lane >> 4;
    const int l15  = lane & 15;

    const int qt = blockIdx.x & 15;          // T/128 = 16 query tiles
    const int bh = blockIdx.x >> 4;          // 0..47
    const int b  = bh / Hn;
    const int h  = bh % Hn;
    const int qbase = qt << 7;
    const int myq   = qbase + wave * 16;
    const float scale = 0.125f;              // 1/sqrt(64)

    const int S3C = 3 * Cdim;

    // K staging map: one b128 async per thread
    const int kr  = tid >> 3;                // 0..31
    const int kc8 = (tid & 7) << 3;          // 0..56
    // V staging map: row-pairs, 4-col chunks -> packed b32 transposed stores
    const int vrp = tid & 15;                // key row pair
    const int vc4 = (tid >> 4) << 2;         // 0..60

    // Q fragments (resident): rows myq + l15, two D-steps of 32
    F16frag qf[2];
    {
        const bf16* qg = qkv + ((size_t)b * Tlen + myq + l15) * S3C + h * Dh;
        #pragma unroll
        for (int ds = 0; ds < 2; ++ds) {
            qf[ds].h[0] = *(const v8bf*)(qg + ds * 32 + hi * 8);
            qf[ds].h[1] = *(const v8bf*)(qg + ds * 32 + 16 + hi * 8);
        }
    }

    v8f oacc[4];
    #pragma unroll
    for (int d = 0; d < 4; ++d) oacc[d] = {};
    float mrow[8], lrow[8];
    #pragma unroll
    for (int v = 0; v < 8; ++v) { mrow[v] = -1.0e30f; lrow[v] = 0.0f; }

    const bf16* kBase = qkv + (size_t)b * Tlen * S3C + Cdim + h * Dh;

    // prologue: async-stage K tile 0
    async_b128(&Klds[0][kr][kc8], kBase + (size_t)kr * S3C + kc8);

    const int nIt = (qbase + 128) >> 5;
    for (int it = 0; it < nIt; ++it) {
        const int kb  = it << 5;
        const int buf = it & 1;

        // prefetch V tile into regs (two adjacent key rows)
        const bf16* vg = qkv + ((size_t)b * Tlen + kb + 2 * vrp) * S3C + 2 * Cdim + h * Dh + vc4;
        v4bf v0 = *(const v4bf*)vg;
        v4bf v1 = *(const v4bf*)(vg + S3C);

        __syncthreads();   // previous tile done: Vlds and Klds[buf^1] free

        if (it + 1 < nIt)  // pipeline next K tile
            async_b128(&Klds[buf ^ 1][kr][kc8],
                       kBase + (size_t)(kb + 32 + kr) * S3C + kc8);

        #pragma unroll
        for (int j = 0; j < 4; ++j) {
            Pack2 p; p.h[0] = v0[j]; p.h[1] = v1[j];
            *(unsigned*)&Vlds[vc4 + j][2 * vrp] = p.u;
        }

        if (it + 1 < nIt) WAIT_ASYNC(1);
        else              WAIT_ASYNC(0);
        __syncthreads();

        // S = Q K^T : two 16x16 key tiles, two D-steps each
        v8f s0 = {}, s1 = {};
        #pragma unroll
        for (int ds = 0; ds < 2; ++ds) {
            F16frag kf0, kf1;
            const bf16* kp0 = &Klds[buf][l15][ds * 32];
            kf0.h[0] = *(const v8bf*)(kp0 + hi * 8);
            kf0.h[1] = *(const v8bf*)(kp0 + 16 + hi * 8);
            const bf16* kp1 = &Klds[buf][16 + l15][ds * 32];
            kf1.h[0] = *(const v8bf*)(kp1 + hi * 8);
            kf1.h[1] = *(const v8bf*)(kp1 + 16 + hi * 8);
            s0 = __builtin_amdgcn_wmma_f32_16x16x32_bf16(false, qf[ds].v, false, kf0.v,
                                                         (short)0, s0, false, false);
            s1 = __builtin_amdgcn_wmma_f32_16x16x32_bf16(false, qf[ds].v, false, kf1.v,
                                                         (short)0, s1, false, false);
        }

        // online softmax; row v+hi*8 lives in one 16-lane half
        #pragma unroll
        for (int v = 0; v < 8; ++v) {
            const int r_ = v + hi * 8;
            const int qi = myq + r_;
            float x0 = (kb + l15      <= qi) ? s0[v] * scale : -3.0e38f;
            float x1 = (kb + 16 + l15 <= qi) ? s1[v] * scale : -3.0e38f;
            float mx = fmaxf(x0, x1);
            mx = fmaxf(mx, __shfl_xor(mx, 1, 32));
            mx = fmaxf(mx, __shfl_xor(mx, 2, 32));
            mx = fmaxf(mx, __shfl_xor(mx, 4, 32));
            mx = fmaxf(mx, __shfl_xor(mx, 8, 32));
            const float mnew  = fmaxf(mrow[v], mx);
            const float alpha = __expf(mrow[v] - mnew);
            const float p0 = __expf(x0 - mnew);
            const float p1 = __expf(x1 - mnew);
            float sum = p0 + p1;
            sum += __shfl_xor(sum, 1, 32);
            sum += __shfl_xor(sum, 2, 32);
            sum += __shfl_xor(sum, 4, 32);
            sum += __shfl_xor(sum, 8, 32);
            lrow[v] = lrow[v] * alpha + sum;
            mrow[v] = mnew;
            #pragma unroll
            for (int db = 0; db < 4; ++db) oacc[db][v] *= alpha;
            Plds[wave][r_][l15]      = (bf16)p0;
            Plds[wave][r_][16 + l15] = (bf16)p1;
        }

        // O += P V  (P: 16x32 wave-private LDS; V: [d][kt] transposed)
        F16frag pf;
        const bf16* pp = &Plds[wave][l15][0];
        pf.h[0] = *(const v8bf*)(pp + hi * 8);
        pf.h[1] = *(const v8bf*)(pp + 16 + hi * 8);
        #pragma unroll
        for (int db = 0; db < 4; ++db) {
            F16frag vf;
            const bf16* vp = &Vlds[db * 16 + l15][0];
            vf.h[0] = *(const v8bf*)(vp + hi * 8);
            vf.h[1] = *(const v8bf*)(vp + 16 + hi * 8);
            oacc[db] = __builtin_amdgcn_wmma_f32_16x16x32_bf16(false, pf.v, false, vf.v,
                                                               (short)0, oacc[db], false, false);
        }
    }

    // normalize and write y (bf16) at [b, t, h*64+d]
    #pragma unroll
    for (int db = 0; db < 4; ++db) {
        const int d = db * 16 + l15;
        #pragma unroll
        for (int v = 0; v < 8; ++v) {
            const int t = myq + v + hi * 8;
            const float val = oacc[db][v] / lrow[v];
            yout[((size_t)b * Tlen + t) * Cdim + h * Dh + d] = (bf16)val;
        }
    }
}

// ---------------------------------------------------------------------------
// host launcher
// ---------------------------------------------------------------------------
extern "C" void kernel_launch(void* const* d_in, const int* in_sizes, int n_in,
                              void* d_out, int out_size, void* d_ws, size_t ws_size,
                              hipStream_t stream) {
    const float* x      = (const float*)d_in[0];   // [B,T,C]
    const float* W_attn = (const float*)d_in[1];   // [C,3C]
    const float* b_attn = (const float*)d_in[2];   // [3C]
    const float* W_proj = (const float*)d_in[3];   // [C,C]
    const float* b_proj = (const float*)d_in[4];   // [C]
    float* out = (float*)d_out;

    const int M  = Bsz * Tlen;       // 8192
    const int K  = Cdim;             // 768
    const int N1 = 3 * Cdim;         // 2304
    const int N2 = Cdim;             // 768

    char* ws = (char*)d_ws;
    bf16* xb   = (bf16*)(ws);                               // 12,582,912 B
    bf16* Wab  = (bf16*)(ws + 12582912);                    //  3,538,944 B
    bf16* Wpb  = (bf16*)(ws + 16121856);                    //  1,179,648 B
    bf16* qkvb = (bf16*)(ws + 17301504);                    // 37,748,736 B
    bf16* yb   = (bf16*)(ws + 55050240);                    // 12,582,912 B

    // 1) convert inputs to bf16
    cvt_bf16_kernel<<<(M * K) / 4 / 256, 256, 0, stream>>>(x, xb, (M * K) / 4);
    cvt_bf16_kernel<<<(K * N1) / 4 / 256, 256, 0, stream>>>(W_attn, Wab, (K * N1) / 4);
    cvt_bf16_kernel<<<(K * N2) / 4 / 256, 256, 0, stream>>>(W_proj, Wpb, (K * N2) / 4);

    // 2) QKV projection (bf16 out)
    gemm_bf16_kernel<true><<<(M / 128) * (N1 / 128), 256, 0, stream>>>(
        xb, Wab, b_attn, (void*)qkvb, M, N1, K);

    // 3) causal flash attention -> y (bf16)
    attn_kernel<<<Bsz * Hn * (Tlen / 128), 256, 0, stream>>>(qkvb, yb);

    // 4) output projection (f32 out)
    gemm_bf16_kernel<false><<<(M / 128) * (N2 / 128), 256, 0, stream>>>(
        yb, Wpb, b_proj, (void*)out, M, N2, K);
}
